// FastDLLMv2Attention_28458453303845
// MI455X (gfx1250) — matrix operand order repaired and checked by reading, not compiled, not run on hardware
//
#include <hip/hip_runtime.h>
#include <hip/hip_bf16.h>

// ---------------------------------------------------------------- types
typedef __attribute__((ext_vector_type(16))) __bf16 bf16x16;
typedef __attribute__((ext_vector_type(8)))  __bf16 bf16x8;
typedef __attribute__((ext_vector_type(8)))  float  f32x8;
typedef unsigned short u16;

#define NUM_HEADS    12
#define NUM_KV_HEADS 2
#define HEAD_DIM     128
#define HIDDEN       1536
#define SEQ          2048
#define BATCH        2
#define CACHE_LEN    2048
#define TOTAL_KEYS   4096          // CACHE_LEN + SEQ
#define MAXC         8192
#define NQKV         2048          // 1536 (q) + 256 (k) + 256 (v) fused columns
#define MTOT         4096          // BATCH*SEQ

// ---------------------------------------------------------------- helpers
static __device__ inline u16 f2bf(float f) {
  unsigned int u = __float_as_uint(f);
  u = (u + 0x7FFFu + ((u >> 16) & 1u)) >> 16;   // round-to-nearest-even
  return (u16)u;
}

static __device__ inline bf16x16 ld_frag(const u16* p0, const u16* p1) {
  bf16x8 lo = __builtin_bit_cast(bf16x8, *(const uint4*)p0);
  bf16x8 hi = __builtin_bit_cast(bf16x8, *(const uint4*)p1);
  return __builtin_shufflevector(lo, hi, 0,1,2,3,4,5,6,7,8,9,10,11,12,13,14,15);
}

static __device__ inline f32x8 wmma_bf16(bf16x16 a, bf16x16 b, f32x8 c) {
  return __builtin_amdgcn_wmma_f32_16x16x32_bf16(false, a, false, b, (short)0, c,
                                                 false, false);
}

// LDS byte offset of a __shared__ object (addrspace(3) pointer value)
static __device__ inline unsigned ldsOffset(const void* p) {
  return (unsigned)(unsigned long long)
      (const __attribute__((address_space(3))) void*)p;
}

// CDNA5 async DMA: memory -> LDS, 16B per lane, tracked by ASYNCcnt
static __device__ inline void async_b128_to_lds(unsigned loff, const void* g) {
  asm volatile("global_load_async_to_lds_b128 %0, %1, off"
               : : "v"(loff), "v"(g) : "memory");
}
static __device__ inline void wait_async0() {
  asm volatile("s_wait_asynccnt 0x0" ::: "memory");
}

// ---------------------------------------------------------------- K0: converters
__global__ __launch_bounds__(256) void cvt_f32_bf16_kernel(const float* __restrict__ x,
                                                           u16* __restrict__ y, int n) {
  int i = blockIdx.x * 256 + threadIdx.x;
  if (i < n) y[i] = f2bf(x[i]);
}

__global__ __launch_bounds__(256) void pack_wqkv_kernel(const float* __restrict__ Wq,
                                                        const float* __restrict__ Wk,
                                                        const float* __restrict__ Wv,
                                                        u16* __restrict__ w) {
  int i = blockIdx.x * 256 + threadIdx.x;           // over 1536*2048
  if (i >= HIDDEN * NQKV) return;
  int k = i >> 11, n = i & (NQKV - 1);
  float v;
  if (n < 1536)       v = Wq[k * 1536 + n];
  else if (n < 1792)  v = Wk[k * 256 + (n - 1536)];
  else                v = Wv[k * 256 + (n - 1792)];
  w[i] = f2bf(v);
}

__global__ __launch_bounds__(256) void copy_cache_kernel(const float* __restrict__ kc,
                                                         const float* __restrict__ vc,
                                                         float* __restrict__ outK,
                                                         float* __restrict__ outV,
                                                         u16* __restrict__ kB,
                                                         u16* __restrict__ vB) {
  int i = blockIdx.x * 256 + threadIdx.x;           // over 2*2*2048*128
  if (i >= BATCH * NUM_KV_HEADS * CACHE_LEN * HEAD_DIM) return;
  int d   = i & 127;
  int pos = (i >> 7) & (CACHE_LEN - 1);
  int bh  = i >> 18;                                // b*2 + hk
  int src = (bh * MAXC + pos) * HEAD_DIM + d;
  int dst = (bh * TOTAL_KEYS + pos) * HEAD_DIM + d;
  float kv = kc[src], vv = vc[src];
  outK[dst] = kv;  kB[dst] = f2bf(kv);
  outV[dst] = vv;  vB[dst] = f2bf(vv);
}

// ---------------------------------------------------------------- K1: fused QKV GEMM + bias + RoPE
// tile 64x128, 256 threads = 8 waves in 2(m) x 4(n) grid, K step 32
__global__ __launch_bounds__(256) void qkv_rope_kernel(
    const u16* __restrict__ hidB, const u16* __restrict__ wqkv,
    const float* __restrict__ bq, const float* __restrict__ bk,
    const float* __restrict__ bv, const int* __restrict__ positions,
    u16* __restrict__ qB, u16* __restrict__ kB, u16* __restrict__ vB,
    float* __restrict__ outK, float* __restrict__ outV) {
  __shared__ u16  As[64 * 40];       // [m][k] row-major, pad to 40
  __shared__ u16  Bs[128 * 40];      // [n][k] n-major (k contiguous)
  __shared__ float Ct[64 * 132];     // f32 epilogue tile

  const int tid  = threadIdx.x;
  const int m0   = blockIdx.y * 64;
  const int n0   = blockIdx.x * 128;
  const int wave = tid >> 5, lane = tid & 31;
  const int half = lane >> 4, lr = lane & 15;
  const int wm = wave >> 2, wn = wave & 3;

  f32x8 acc[2][2] = {};

  for (int k0 = 0; k0 < HIDDEN; k0 += 32) {
    { // stage A: 64x32 via async DMA (memory -> LDS, ASYNCcnt)
      int m = tid >> 2, kc = (tid & 3) << 3;
      const u16* gp = &hidB[(m0 + m) * HIDDEN + k0 + kc];
      async_b128_to_lds(ldsOffset(&As[m * 40 + kc]), gp);
      if (k0 + 32 < HIDDEN) __builtin_prefetch(gp + 32, 0, 1);
    }
    { // stage B transposed into [n][k]: needs VGPR round-trip (element shuffle)
      int kp = (tid & 15) << 1;          // k, k+1
      int nb = (tid >> 4) << 3;          // 8 columns
      uint4 r0 = *(const uint4*)&wqkv[(k0 + kp)     * NQKV + n0 + nb];
      uint4 r1 = *(const uint4*)&wqkv[(k0 + kp + 1) * NQKV + n0 + nb];
      const u16* e0 = (const u16*)&r0;
      const u16* e1 = (const u16*)&r1;
      #pragma unroll
      for (int i = 0; i < 8; ++i) {
        unsigned int pk = (unsigned int)e0[i] | ((unsigned int)e1[i] << 16);
        *(unsigned int*)&Bs[(nb + i) * 40 + kp] = pk;
      }
    }
    wait_async0();
    __syncthreads();
    bf16x16 bfr[2];
    #pragma unroll
    for (int ns = 0; ns < 2; ++ns) {
      int nr = (wn * 32 + ns * 16 + lr) * 40;
      bfr[ns] = ld_frag(&Bs[nr + half * 8], &Bs[nr + 16 + half * 8]);
    }
    #pragma unroll
    for (int ms = 0; ms < 2; ++ms) {
      int mr = (wm * 32 + ms * 16 + lr) * 40;
      bf16x16 a = ld_frag(&As[mr + half * 8], &As[mr + 16 + half * 8]);
      #pragma unroll
      for (int ns = 0; ns < 2; ++ns)
        acc[ms][ns] = wmma_bf16(a, bfr[ns], acc[ms][ns]);
    }
    __syncthreads();
  }

  // spill accumulators to LDS (C layout: M = v + 8*half, N = lane%16)
  #pragma unroll
  for (int ms = 0; ms < 2; ++ms)
    #pragma unroll
    for (int ns = 0; ns < 2; ++ns)
      #pragma unroll
      for (int v = 0; v < 8; ++v)
        Ct[(wm * 32 + ms * 16 + v + 8 * half) * 132 + wn * 32 + ns * 16 + lr] =
            acc[ms][ns][v];
  __syncthreads();

  const int cb = n0 >> 7;   // 0..11 q heads, 12..13 k heads, 14..15 v heads
  for (int i = tid; i < 64 * 128; i += 256) {
    int m = i >> 7, d = i & 127;
    int gm = m0 + m;
    int b = gm >> 11, s = gm & (SEQ - 1);
    int n = n0 + d;
    float bias = (cb < 12) ? bq[n] : (cb < 14 ? bk[n - 1536] : bv[n - 1792]);
    float x = Ct[m * 132 + d] + bias;
    if (cb < 14) {                         // RoPE on q and k
      int dp = d ^ 64;
      int np = n0 + dp;
      float biasp = (cb < 12) ? bq[np] : bk[np - 1536];
      float xp = Ct[m * 132 + dp] + biasp;
      float p = (float)positions[b * SEQ + s];
      float invf = __powf(1.0e6f, -(float)(d & 63) * (1.0f / 64.0f));
      float sn, cs;
      __sincosf(p * invf, &sn, &cs);
      x = (d < 64) ? (x * cs - xp * sn) : (x * cs + xp * sn);
      if (cb < 12) {
        qB[((b * NUM_HEADS + cb) * SEQ + s) * HEAD_DIM + d] = f2bf(x);
      } else {
        int idx = ((b * NUM_KV_HEADS + (cb - 12)) * TOTAL_KEYS + CACHE_LEN + s) *
                      HEAD_DIM + d;
        outK[idx] = x;
        kB[idx]   = f2bf(x);
      }
    } else {
      int idx = ((b * NUM_KV_HEADS + (cb - 14)) * TOTAL_KEYS + CACHE_LEN + s) *
                    HEAD_DIM + d;
      outV[idx] = x;
      vB[idx]   = f2bf(x);
    }
  }
}

// ---------------------------------------------------------------- K2: flash attention
// 128 threads = 4 waves; block owns 64 query rows of one (b,h); each wave owns 16 rows
// and the full key-block width so softmax reductions are wave-local.
__global__ __launch_bounds__(128) void attn_kernel(
    const u16* __restrict__ qB, const u16* __restrict__ kB,
    const u16* __restrict__ vB, const float* __restrict__ mask,
    u16* __restrict__ aB) {
  __shared__ u16 Qs[64 * 136];
  __shared__ u16 Ks[64 * 136];      // [key][d] — already B-fragment friendly
  __shared__ u16 Vt[128 * 72];      // [d][key] transposed
  __shared__ u16 Ps[64 * 72];       // probabilities, [m][key]

  const int tid = threadIdx.x;
  const int bh = blockIdx.y;                  // 0..23
  const int b = bh / NUM_HEADS, h = bh % NUM_HEADS;
  const int hk = h / (NUM_HEADS / NUM_KV_HEADS);
  const int m0 = blockIdx.x * 64;
  const int wave = tid >> 5, lane = tid & 31;
  const int half = lane >> 4, lr = lane & 15;
  const int mrow = wave * 16 + lr;
  const float scaling = 0.08838834764831845f;   // 128^-0.5

  { // stage Q tile 64x128 via async DMA
    const u16* qsrc = qB + ((b * NUM_HEADS + h) * SEQ + m0) * HEAD_DIM;
    #pragma unroll
    for (int j = 0; j < 8; ++j) {
      int fi = tid + j * 128;
      int r = fi >> 4, c8 = (fi & 15) << 3;
      async_b128_to_lds(ldsOffset(&Qs[r * 136 + c8]), &qsrc[r * HEAD_DIM + c8]);
    }
  }
  wait_async0();
  __syncthreads();
  bf16x16 qf[4];
  #pragma unroll
  for (int kk = 0; kk < 4; ++kk)
    qf[kk] = ld_frag(&Qs[mrow * 136 + kk * 32 + half * 8],
                     &Qs[mrow * 136 + kk * 32 + 16 + half * 8]);

  f32x8 oacc[8] = {};
  float rmax[8], rsum[8];
  #pragma unroll
  for (int v = 0; v < 8; ++v) { rmax[v] = -1e30f; rsum[v] = 0.0f; }

  const u16* ksrc = kB + (b * NUM_KV_HEADS + hk) * TOTAL_KEYS * HEAD_DIM;
  const u16* vsrc = vB + (b * NUM_KV_HEADS + hk) * TOTAL_KEYS * HEAD_DIM;
  const float* msrc =
      mask + (size_t)((b * NUM_HEADS + h) * SEQ + m0) * TOTAL_KEYS;

  for (int kbp = 0; kbp < TOTAL_KEYS; kbp += 64) {
    __syncthreads();
    #pragma unroll
    for (int j = 0; j < 8; ++j) {         // stage K block [key][d] via async DMA
      int fi = tid + j * 128;
      int r = fi >> 4, c8 = (fi & 15) << 3;
      async_b128_to_lds(ldsOffset(&Ks[r * 136 + c8]),
                        &ksrc[(kbp + r) * HEAD_DIM + c8]);
    }
    #pragma unroll
    for (int j = 0; j < 8; ++j) {         // stage V block transposed [d][key]
      int fi = tid + j * 128;
      int r = fi >> 4, c8 = (fi & 15) << 3;
      uint4 vv = *(const uint4*)&vsrc[(kbp + r) * HEAD_DIM + c8];
      const u16* e = (const u16*)&vv;
      #pragma unroll
      for (int i = 0; i < 8; ++i) Vt[(c8 + i) * 72 + r] = e[i];
    }
    wait_async0();
    __syncthreads();

    // scores = q @ k^T
    f32x8 sacc[4] = {};
    #pragma unroll
    for (int ns = 0; ns < 4; ++ns) {
      int nr = (ns * 16 + lr) * 136;
      #pragma unroll
      for (int kk = 0; kk < 4; ++kk) {
        bf16x16 bf = ld_frag(&Ks[nr + kk * 32 + half * 8],
                             &Ks[nr + kk * 32 + 16 + half * 8]);
        sacc[ns] = wmma_bf16(qf[kk], bf, sacc[ns]);
      }
    }
    // scale + additive mask (streamed non-temporally: 805MB, read once)
    #pragma unroll
    for (int ns = 0; ns < 4; ++ns)
      #pragma unroll
      for (int v = 0; v < 8; ++v) {
        int mrl = wave * 16 + v + 8 * half;
        sacc[ns][v] = sacc[ns][v] * scaling +
            __builtin_nontemporal_load(&msrc[mrl * TOTAL_KEYS + kbp + ns * 16 + lr]);
      }
    // online softmax
    #pragma unroll
    for (int v = 0; v < 8; ++v) {
      float mx = fmaxf(fmaxf(sacc[0][v], sacc[1][v]),
                       fmaxf(sacc[2][v], sacc[3][v]));
      #pragma unroll
      for (int off = 1; off < 16; off <<= 1)
        mx = fmaxf(mx, __shfl_xor(mx, off, 32));
      float mnew = fmaxf(rmax[v], mx);
      float corr = __expf(rmax[v] - mnew);
      rmax[v] = mnew;
      rsum[v] *= corr;
      #pragma unroll
      for (int n = 0; n < 8; ++n) oacc[n][v] *= corr;
    }
    #pragma unroll
    for (int ns = 0; ns < 4; ++ns)
      #pragma unroll
      for (int v = 0; v < 8; ++v) {
        float pv = __expf(sacc[ns][v] - rmax[v]);
        sacc[ns][v] = pv;
        Ps[(wave * 16 + v + 8 * half) * 72 + ns * 16 + lr] = f2bf(pv);
      }
    #pragma unroll
    for (int v = 0; v < 8; ++v) {
      float sm = sacc[0][v] + sacc[1][v] + sacc[2][v] + sacc[3][v];
      #pragma unroll
      for (int off = 1; off < 16; off <<= 1)
        sm += __shfl_xor(sm, off, 32);
      rsum[v] += sm;
    }
    // out += P @ V   (intra-wave LDS ops are in-order, Ps is wave-private rows)
    #pragma unroll
    for (int kc = 0; kc < 2; ++kc) {
      bf16x16 pa = ld_frag(&Ps[mrow * 72 + kc * 32 + half * 8],
                           &Ps[mrow * 72 + kc * 32 + 16 + half * 8]);
      #pragma unroll
      for (int n = 0; n < 8; ++n) {
        int nr = (n * 16 + lr) * 72;
        bf16x16 vb = ld_frag(&Vt[nr + kc * 32 + half * 8],
                             &Vt[nr + kc * 32 + 16 + half * 8]);
        oacc[n] = wmma_bf16(pa, vb, oacc[n]);
      }
    }
  }

  // normalize and store (B,S,12*128) bf16 for the out-projection GEMM
  u16* dst = aB + (b * SEQ + m0) * HIDDEN + h * HEAD_DIM;
  #pragma unroll
  for (int v = 0; v < 8; ++v) {
    int mrl = wave * 16 + v + 8 * half;
    float inv = 1.0f / rsum[v];
    #pragma unroll
    for (int n = 0; n < 8; ++n)
      dst[mrl * HIDDEN + n * 16 + lr] = f2bf(oacc[n][v] * inv);
  }
}

// ---------------------------------------------------------------- K3: output projection
__global__ __launch_bounds__(256) void out_proj_kernel(const u16* __restrict__ aB,
                                                       const u16* __restrict__ woB,
                                                       float* __restrict__ out) {
  __shared__ u16 As[64 * 40];
  __shared__ u16 Bs[128 * 40];
  const int tid = threadIdx.x;
  const int m0 = blockIdx.y * 64;
  const int n0 = blockIdx.x * 128;
  const int wave = tid >> 5, lane = tid & 31;
  const int half = lane >> 4, lr = lane & 15;
  const int wm = wave >> 2, wn = wave & 3;

  f32x8 acc[2][2] = {};
  for (int k0 = 0; k0 < HIDDEN; k0 += 32) {
    { // A tile via async DMA
      int m = tid >> 2, kc = (tid & 3) << 3;
      const u16* gp = &aB[(m0 + m) * HIDDEN + k0 + kc];
      async_b128_to_lds(ldsOffset(&As[m * 40 + kc]), gp);
      if (k0 + 32 < HIDDEN) __builtin_prefetch(gp + 32, 0, 1);
    }
    {
      int kp = (tid & 15) << 1;
      int nb = (tid >> 4) << 3;
      uint4 r0 = *(const uint4*)&woB[(k0 + kp)     * HIDDEN + n0 + nb];
      uint4 r1 = *(const uint4*)&woB[(k0 + kp + 1) * HIDDEN + n0 + nb];
      const u16* e0 = (const u16*)&r0;
      const u16* e1 = (const u16*)&r1;
      #pragma unroll
      for (int i = 0; i < 8; ++i) {
        unsigned int pk = (unsigned int)e0[i] | ((unsigned int)e1[i] << 16);
        *(unsigned int*)&Bs[(nb + i) * 40 + kp] = pk;
      }
    }
    wait_async0();
    __syncthreads();
    bf16x16 bfr[2];
    #pragma unroll
    for (int ns = 0; ns < 2; ++ns) {
      int nr = (wn * 32 + ns * 16 + lr) * 40;
      bfr[ns] = ld_frag(&Bs[nr + half * 8], &Bs[nr + 16 + half * 8]);
    }
    #pragma unroll
    for (int ms = 0; ms < 2; ++ms) {
      int mr = (wm * 32 + ms * 16 + lr) * 40;
      bf16x16 a = ld_frag(&As[mr + half * 8], &As[mr + 16 + half * 8]);
      #pragma unroll
      for (int ns = 0; ns < 2; ++ns)
        acc[ms][ns] = wmma_bf16(a, bfr[ns], acc[ms][ns]);
    }
    __syncthreads();
  }
  #pragma unroll
  for (int ms = 0; ms < 2; ++ms)
    #pragma unroll
    for (int ns = 0; ns < 2; ++ns)
      #pragma unroll
      for (int v = 0; v < 8; ++v) {
        int gm = m0 + wm * 32 + ms * 16 + v + 8 * half;
        int gn = n0 + wn * 32 + ns * 16 + lr;
        out[gm * HIDDEN + gn] = acc[ms][ns][v];
      }
}

// ---------------------------------------------------------------- launcher
extern "C" void kernel_launch(void* const* d_in, const int* in_sizes, int n_in,
                              void* d_out, int out_size, void* d_ws, size_t ws_size,
                              hipStream_t stream) {
  (void)in_sizes; (void)n_in; (void)out_size; (void)ws_size;
  const int*   positions = (const int*)  d_in[0];
  const float* hidden    = (const float*)d_in[1];
  const float* k_cache   = (const float*)d_in[2];
  const float* v_cache   = (const float*)d_in[3];
  const float* mask      = (const float*)d_in[6];
  const float* Wq        = (const float*)d_in[7];
  const float* bq        = (const float*)d_in[8];
  const float* Wk        = (const float*)d_in[9];
  const float* bk        = (const float*)d_in[10];
  const float* Wv        = (const float*)d_in[11];
  const float* bv        = (const float*)d_in[12];
  const float* Wo        = (const float*)d_in[13];

  float* out   = (float*)d_out;                       // (B,S,1536)
  float* new_k = out + (size_t)MTOT * HIDDEN;         // (B,2,4096,128)
  float* new_v = new_k + (size_t)BATCH * NUM_KV_HEADS * TOTAL_KEYS * HEAD_DIM;

  // workspace carve (bf16/u16 buffers)
  char* ws = (char*)d_ws;
  u16* hidB  = (u16*)ws;                 ws += (size_t)MTOT * HIDDEN * 2;
  u16* wqkvB = (u16*)ws;                 ws += (size_t)HIDDEN * NQKV * 2;
  u16* woB   = (u16*)ws;                 ws += (size_t)HIDDEN * HIDDEN * 2;
  u16* qB    = (u16*)ws;                 ws += (size_t)BATCH * NUM_HEADS * SEQ * HEAD_DIM * 2;
  u16* kB    = (u16*)ws;                 ws += (size_t)BATCH * NUM_KV_HEADS * TOTAL_KEYS * HEAD_DIM * 2;
  u16* vB    = (u16*)ws;                 ws += (size_t)BATCH * NUM_KV_HEADS * TOTAL_KEYS * HEAD_DIM * 2;
  u16* aB    = (u16*)ws;                 ws += (size_t)MTOT * HIDDEN * 2;

  int nHid = MTOT * HIDDEN;
  cvt_f32_bf16_kernel<<<(nHid + 255) / 256, 256, 0, stream>>>(hidden, hidB, nHid);
  int nWq = HIDDEN * NQKV;
  pack_wqkv_kernel<<<(nWq + 255) / 256, 256, 0, stream>>>(Wq, Wk, Wv, wqkvB);
  int nWo = HIDDEN * HIDDEN;
  cvt_f32_bf16_kernel<<<(nWo + 255) / 256, 256, 0, stream>>>(Wo, woB, nWo);
  int nCache = BATCH * NUM_KV_HEADS * CACHE_LEN * HEAD_DIM;
  copy_cache_kernel<<<(nCache + 255) / 256, 256, 0, stream>>>(k_cache, v_cache,
                                                              new_k, new_v, kB, vB);

  qkv_rope_kernel<<<dim3(NQKV / 128, MTOT / 64), 256, 0, stream>>>(
      hidB, wqkvB, bq, bk, bv, positions, qB, kB, vB, new_k, new_v);

  attn_kernel<<<dim3(SEQ / 64, BATCH * NUM_HEADS), 128, 0, stream>>>(
      qB, kB, vB, mask, aB);

  out_proj_kernel<<<dim3(HIDDEN / 128, MTOT / 64), 256, 0, stream>>>(aB, woB, out);
}
